// TinySSD_30133490549546
// MI455X (gfx1250) — matrix operand order repaired and checked by reading, not compiled, not run on hardware
//
#include <hip/hip_runtime.h>

// ---------------- CDNA5 WMMA types ----------------
typedef __attribute__((ext_vector_type(16))) _Float16 v16h;
typedef __attribute__((ext_vector_type(8)))  _Float16 v8h;
typedef __attribute__((ext_vector_type(8)))  float    v8f;

#define WAVES 4
#define TBLOCK (WAVES * 32)
#define TILES 4            // N-tiles (16 positions each) per wave

// =====================================================================
// 3x3 conv (pad=1) as 9 shifted GEMM taps on v_wmma_f32_16x16x32_f16.
// Activations NHWC f16, channels padded to Cp (mult of 16).
// K ordering: k = tap*Cp + c  (tap = ky*3+kx). Kp = roundup(9*Cp, 32).
// Weights prepacked f16 [Coutp][Kp]; 16*MT-row tile staged in LDS per block.
// Each wave computes a (16*MT Cout) x 64(pos) tile: each B fragment feeds
// MT WMMAs, each A fragment feeds TILES WMMAs -> MT*TILES WMMAs per K-step.
// Out-of-bounds B loads are redirected to a zeroed scratch page via a
// branchless pointer select (v_cndmask) -> unconditional b128 loads.
// =====================================================================
template <int MT>
__global__ __launch_bounds__(TBLOCK)
void conv3x3_wmma(const _Float16* __restrict__ in,   // [B][H][W][Cp] f16
                  const _Float16* __restrict__ wA,   // [Coutp][Kp]   f16
                  const float*    __restrict__ scale,// [Cout]
                  const float*    __restrict__ shift,// [Cout]
                  _Float16* __restrict__ outH,       // NHWC f16 [B][H][W][CoutP]
                  float*    __restrict__ outF,       // NHWC f32 into d_out
                  const _Float16* __restrict__ zbuf, // 64 zero halves
                  int Cp, int lgCp, int Cout, int CoutP,
                  int H, int W, int relu, int bStrideF, int stageOffF)
{
    const int Kp = (9 * Cp + 31) & ~31;
    const int HW = H * W;
    extern __shared__ _Float16 smem[];            // [16*MT][Kp] weight tile

    const int tid  = threadIdx.x;
    const int lane = tid & 31;
    const int wave = tid >> 5;
    const int coutBase = blockIdx.x * (16 * MT);
    const int pBase    = (blockIdx.y * WAVES + wave) * (16 * TILES);
    const int bb       = blockIdx.z;

    // ---- cooperative global->LDS copy of the (16*MT) x Kp weight tile ----
    {
        const v8h* src = (const v8h*)(wA + (size_t)coutBase * Kp);
        v8h*       dst = (v8h*)smem;
        const int nChunks = (16 * MT * Kp) / 8;
        for (int i = tid; i < nChunks; i += TBLOCK) dst[i] = src[i];
    }
    __syncthreads();

    const int n    = lane & 15;
    const int hi   = (lane & 16) >> 1;  // +8 C-row for upper lanes
    const int mA   = lane & 15;         // A-matrix row held by this lane
    const int kSel = lane & 16;         // B-frag K offset for upper lanes

    int ps[TILES], ys[TILES], xs[TILES];
    bool oks[TILES];
    ptrdiff_t bases[TILES];
#pragma unroll
    for (int t = 0; t < TILES; ++t) {
        const int p = pBase + 16 * t + n;
        ps[t]  = p;
        const bool ok = (p < HW);
        oks[t] = ok;
        const int y = ok ? p / W : 0;
        const int x = ok ? p % W : 0;
        ys[t] = y; xs[t] = x;
        bases[t] = (((ptrdiff_t)bb * H + y) * W + x) * Cp;
    }
    if (oks[0]) __builtin_prefetch(in + bases[0], 0, 1);

    v8f acc[MT][TILES];
#pragma unroll
    for (int mt = 0; mt < MT; ++mt)
#pragma unroll
        for (int t = 0; t < TILES; ++t) acc[mt][t] = (v8f){};

    for (int k0 = 0; k0 < Kp; k0 += 32) {
        // A fragments from LDS (contiguous K: lanes>=16 hold K+16..K+31)
        v16h a[MT];
#pragma unroll
        for (int mt = 0; mt < MT; ++mt) {
            const int klo = (mA + 16 * mt) * Kp + k0 + kSel;
            v8h lo = *(const v8h*)&smem[klo];
            v8h hh = *(const v8h*)&smem[klo + 8];
#pragma unroll
            for (int i = 0; i < 8; ++i) { a[mt][i] = lo[i]; a[mt][i + 8] = hh[i]; }
        }

        // Branchless tap decode: this lane-half's 16-run lies in one tap.
        const int kb  = k0 + kSel;
        const int tap = kb >> lgCp;              // 0..8 (>=9 => zero pad)
        const int c   = kb & (Cp - 1);
        const int ty  = (tap * 11) >> 5;         // tap / 3 (exact for 0..29)
        const int tx  = tap - ty * 3;            // tap % 3
        const bool tOk = (tap < 9);
        const ptrdiff_t doff = ((ptrdiff_t)(ty - 1) * W + (tx - 1)) * Cp + c;

#pragma unroll
        for (int t = 0; t < TILES; ++t) {
            const int yy = ys[t] + ty - 1;
            const int xx = xs[t] + tx - 1;
            const bool ok = oks[t] && tOk &&
                            ((unsigned)yy < (unsigned)H) &&
                            ((unsigned)xx < (unsigned)W);
            const _Float16* src = ok ? (in + bases[t] + doff) : zbuf;
            v8h lo = *(const v8h*)src;           // unconditional b128
            v8h hh = *(const v8h*)(src + 8);     // unconditional b128
            v16h bf;
#pragma unroll
            for (int i = 0; i < 8; ++i) { bf[i] = lo[i]; bf[i + 8] = hh[i]; }
#pragma unroll
            for (int mt = 0; mt < MT; ++mt)      // B fragment feeds MT WMMAs
                acc[mt][t] = __builtin_amdgcn_wmma_f32_16x16x32_f16(
                                 false, a[mt], false, bf, (short)0,
                                 acc[mt][t], false, false);
        }
    }

    // ---- epilogue: lane holds 8 contiguous channels per sub-tile ----
#pragma unroll
    for (int mt = 0; mt < MT; ++mt) {
        const int mBase = coutBase + 16 * mt + hi;
#pragma unroll
        for (int t = 0; t < TILES; ++t) {
            if (!oks[t]) continue;
            const int p = ps[t];
            if (outH) {
                v8h o;
#pragma unroll
                for (int r = 0; r < 8; ++r) {
                    const int m = mBase + r;
                    float v = (m < Cout) ? acc[mt][t][r] * scale[m] + shift[m] : 0.f;
                    if (relu) v = fmaxf(v, 0.f);
                    o[r] = (_Float16)v;
                }
                *(v8h*)&outH[((size_t)bb * HW + p) * CoutP + mBase] = o;
            }
            if (outF) {
#pragma unroll
                for (int r = 0; r < 8; ++r) {
                    const int m = mBase + r;
                    if (m < Cout) {
                        float v = acc[mt][t][r] * scale[m] + shift[m];
                        if (relu) v = fmaxf(v, 0.f);
                        outF[(size_t)bb * bStrideF + stageOffF + (size_t)p * Cout + m] = v;
                    }
                }
            }
        }
    }
}

// ---------------- support kernels ----------------
__global__ void zfill(_Float16* z, int n) {
    int i = blockIdx.x * blockDim.x + threadIdx.x;
    if (i < n) z[i] = (_Float16)0.f;
}

// X f32 NCHW [B][3][HW] -> f16 NHWC padded [B][HW][16]
__global__ void cvt_x_nhwc(const float* __restrict__ X, _Float16* __restrict__ d,
                           int HW, int B) {
    long i = (long)blockIdx.x * blockDim.x + threadIdx.x;   // over B*HW*16
    if (i >= (long)B * HW * 16) return;
    const int c = (int)(i & 15);
    const long ph = i >> 4;
    const int p  = (int)(ph % HW);
    const int bb = (int)(ph / HW);
    d[i] = (c < 3) ? (_Float16)X[((size_t)bb * 3 + c) * HW + p] : (_Float16)0.f;
}

// f32 [Cout][Cin][3][3] -> f16 [Coutp][Kp], k = tap*Cp + c, zero padded
__global__ void prepack_w(const float* __restrict__ w, _Float16* __restrict__ d,
                          int Cout, int Cin, int Cp, int lgCp) {
    const int Kp = (9 * Cp + 31) & ~31, Coutp = (Cout + 15) & ~15;
    long i = (long)blockIdx.x * blockDim.x + threadIdx.x;
    if (i >= (long)Coutp * Kp) return;
    const int k = (int)(i % Kp), m = (int)(i / Kp);
    const int tap = k >> lgCp, c = k & (Cp - 1);
    _Float16 v = (_Float16)0.f;
    if (m < Cout && tap < 9 && c < Cin)
        v = (_Float16)w[((size_t)m * Cin + c) * 9 + tap];
    d[i] = v;
}

__global__ void bn_ss(const float* b, const float* g, const float* be,
                      const float* me, const float* va,
                      float* scale, float* shift, int C) {
    int c = blockIdx.x * blockDim.x + threadIdx.x;
    if (c < C) {
        float inv = g[c] * rsqrtf(va[c] + 1e-5f);
        scale[c] = inv;
        shift[c] = (b[c] - me[c]) * inv + be[c];
    }
}

__global__ void pred_ss(const float* b, float* scale, float* shift, int C) {
    int c = blockIdx.x * blockDim.x + threadIdx.x;
    if (c < C) { scale[c] = 1.f; shift[c] = b[c]; }
}

// 2x2 maxpool, NHWC (channel-contiguous, coalesced)
__global__ void maxpool2(const _Float16* __restrict__ in, _Float16* __restrict__ out,
                         int C, int H, int W) {
    const int Ho = H >> 1, Wo = W >> 1;
    const int bb = blockIdx.z;
    long i = (long)blockIdx.x * blockDim.x + threadIdx.x;   // over Ho*Wo*C
    if (i >= (long)C * Ho * Wo) return;
    const int c  = (int)(i % C);
    const int xo = (int)((i / C) % Wo);
    const int yo = (int)(i / ((long)C * Wo));
    const _Float16* p = in + (((size_t)bb * H + 2 * yo) * W + 2 * xo) * C + c;
    const size_t rs = (size_t)W * C;
    _Float16 m0 = p[0]  > p[C]      ? p[0]  : p[C];
    _Float16 m1 = p[rs] > p[rs + C] ? p[rs] : p[rs + C];
    out[(((size_t)bb * Ho + yo) * Wo + xo) * C + c] = m0 > m1 ? m0 : m1;
}

// global maxpool, NHWC -> [B][C]
__global__ void gmaxpool(const _Float16* __restrict__ in, _Float16* __restrict__ out,
                         int C, int HW, int B) {
    int i = blockIdx.x * blockDim.x + threadIdx.x;   // over B*C
    if (i >= B * C) return;
    const int bb = i / C, c = i % C;
    _Float16 m = in[(size_t)bb * HW * C + c];
    for (int p = 1; p < HW; ++p) {
        _Float16 v = in[((size_t)bb * HW + p) * C + c];
        if (v > m) m = v;
    }
    out[(size_t)bb * C + c] = m;
}

__global__ void anchors_k(float* __restrict__ out, int h, int w, float s0, float s1) {
    int i = blockIdx.x * blockDim.x + threadIdx.x;   // over h*w*4
    if (i >= h * w * 4) return;
    const int a = i & 3, p = i >> 2;
    const int xg = p % w, yg = p / w;
    const float cx = (xg + 0.5f) / w, cy = (yg + 0.5f) / h;
    const float SQ2 = 1.41421356237f;
    float ws_, hs_;
    if      (a == 0) { ws_ = s0;        hs_ = s0;        }
    else if (a == 1) { ws_ = s1;        hs_ = s1;        }
    else if (a == 2) { ws_ = s0 * SQ2;  hs_ = s0 / SQ2;  }
    else             { ws_ = s0 / SQ2;  hs_ = s0 * SQ2;  }
    float* o = out + (size_t)i * 4;
    o[0] = cx - ws_ * 0.5f; o[1] = cy - hs_ * 0.5f;
    o[2] = cx + ws_ * 0.5f; o[3] = cy + hs_ * 0.5f;
}

// =====================================================================
extern "C" void kernel_launch(void* const* d_in, const int* in_sizes, int n_in,
                              void* d_out, int out_size, void* d_ws, size_t ws_size,
                              hipStream_t stream) {
    const int NB = in_sizes[0] / (3 * 512 * 512);

    // ---- gather parameter pointers (dict traversal order) ----
    int gi = 0;
    const float* X = (const float*)d_in[gi++];
    struct CB { const float *w, *b, *g, *be, *me, *va; int cin, cout; };
    static const int cio[12][2] = {{3,16},{16,16},{16,32},{32,32},{32,64},{64,64},
                                   {64,128},{128,128},{128,128},{128,128},{128,128},{128,128}};
    CB cv[12];
    for (int l = 0; l < 12; ++l) {
        cv[l].w  = (const float*)d_in[gi++]; cv[l].b  = (const float*)d_in[gi++];
        cv[l].g  = (const float*)d_in[gi++]; cv[l].be = (const float*)d_in[gi++];
        cv[l].me = (const float*)d_in[gi++]; cv[l].va = (const float*)d_in[gi++];
        cv[l].cin = cio[l][0]; cv[l].cout = cio[l][1];
    }
    const float *clsW[5], *clsB[5], *bbW[5], *bbB[5];
    for (int s = 0; s < 5; ++s) { clsW[s] = (const float*)d_in[gi++]; clsB[s] = (const float*)d_in[gi++]; }
    for (int s = 0; s < 5; ++s) { bbW[s]  = (const float*)d_in[gi++]; bbB[s]  = (const float*)d_in[gi++]; }
    static const int featC[5] = {64, 128, 128, 128, 128};

    // ---- workspace carve ----
    char* wsb = (char*)d_ws;
    size_t off = 0;
    auto carve = [&](size_t bytes) -> void* {
        void* pp = wsb + off; off = (off + bytes + 255) & ~(size_t)255; return pp;
    };
    _Float16* Xp   = (_Float16*)carve((size_t)NB * 512 * 512 * 16 * 2); // NHWC pad 3->16
    _Float16* A    = (_Float16*)carve((size_t)NB * 512 * 512 * 16 * 2);
    _Float16* Bb   = (_Float16*)carve((size_t)NB * 512 * 512 * 16 * 2);
    _Float16* zbuf = (_Float16*)carve(64 * 2);                           // zero page

    zfill<<<dim3(1), dim3(64), 0, stream>>>(zbuf, 64);

    // ---- prepack all 22 conv layers (12 bn + 5 cls + 5 bbox) ----
    _Float16* wp[22]; float* sc[22]; float* sh[22];
    int lCp[22], lLg[22], lCout[22];
    for (int l = 0; l < 22; ++l) {
        int cin, cout; const float *w, *b;
        if (l < 12)      { cin = cv[l].cin; cout = cv[l].cout; w = cv[l].w; b = cv[l].b; }
        else if (l < 17) { cin = featC[l - 12]; cout = 84; w = clsW[l - 12]; b = clsB[l - 12]; }
        else             { cin = featC[l - 17]; cout = 16; w = bbW[l - 17]; b = bbB[l - 17]; }
        int Cp = (cin + 15) & ~15;                       // 3->16, others exact
        int lg = 31 - __builtin_clz((unsigned)Cp);
        lCp[l] = Cp; lLg[l] = lg; lCout[l] = cout;
        const int Kp = (9 * Cp + 31) & ~31, Coutp = (cout + 15) & ~15;
        wp[l] = (_Float16*)carve((size_t)Coutp * Kp * 2);
        sc[l] = (float*)carve((size_t)cout * 4);
        sh[l] = (float*)carve((size_t)cout * 4);
        const long nw = (long)Coutp * Kp;
        prepack_w<<<dim3((unsigned)((nw + 255) / 256)), dim3(256), 0, stream>>>(w, wp[l], cout, cin, Cp, lg);
        if (l < 12)
            bn_ss<<<dim3((cout + 63) / 64), dim3(64), 0, stream>>>(b, cv[l].g, cv[l].be, cv[l].me, cv[l].va, sc[l], sh[l], cout);
        else
            pred_ss<<<dim3((cout + 63) / 64), dim3(64), 0, stream>>>(b, sc[l], sh[l], cout);
    }

    // ---- input cast f32 NCHW -> f16 NHWC padded ----
    {
        const long n = (long)NB * 512 * 512 * 16;
        cvt_x_nhwc<<<dim3((unsigned)((n + 255) / 256)), dim3(256), 0, stream>>>(X, Xp, 512 * 512, NB);
    }

    auto run_conv = [&](const _Float16* in, int l, int H, int W, _Float16* outH,
                        float* outF, int relu, int bStrideF, int stageOffF) {
        const int Kp = (9 * lCp[l] + 31) & ~31;
        const int CoutP = (lCout[l] + 15) & ~15;
        const int posPerBlock = 16 * TILES * WAVES;
        const int gy = (H * W + posPerBlock - 1) / posPerBlock;
        if (lCout[l] >= 32) {   // wide-M path: each B fragment feeds 2 WMMAs
            dim3 grid((lCout[l] + 31) / 32, gy, NB);
            conv3x3_wmma<2><<<grid, dim3(TBLOCK), (size_t)32 * Kp * 2, stream>>>(
                in, wp[l], sc[l], sh[l], outH, outF, zbuf,
                lCp[l], lLg[l], lCout[l], CoutP, H, W, relu, bStrideF, stageOffF);
        } else {
            dim3 grid((lCout[l] + 15) / 16, gy, NB);
            conv3x3_wmma<1><<<grid, dim3(TBLOCK), (size_t)16 * Kp * 2, stream>>>(
                in, wp[l], sc[l], sh[l], outH, outF, zbuf,
                lCp[l], lLg[l], lCout[l], CoutP, H, W, relu, bStrideF, stageOffF);
        }
    };
    auto run_pool = [&](const _Float16* in, _Float16* out, int C, int H, int W) {
        const long n = (long)C * (H / 2) * (W / 2);
        maxpool2<<<dim3((unsigned)((n + 255) / 256), 1, NB), dim3(256), 0, stream>>>(in, out, C, H, W);
    };

    // ---- output regions ----
    float* outF   = (float*)d_out;
    const int ANCH_N  = 21764;                       // 16384+4096+1024+256+4 rows
    float* clsBase = outF + (size_t)ANCH_N * 4;      // 87056
    const int CLS_PB = ANCH_N * 21;                  // 457044
    float* bbBase  = clsBase + (size_t)NB * CLS_PB;
    const int BB_PB = ANCH_N * 4;                    // 87056
    static const int clsOff[5] = {0, 344064, 430080, 451584, 456960};
    static const int bbOff[5]  = {0,  65536,  81920,  86016,  87040};

    auto run_preds = [&](const _Float16* feat, int s, int H, int W) {
        run_conv(feat, 12 + s, H, W, nullptr, clsBase, 0, CLS_PB, clsOff[s]);
        run_conv(feat, 17 + s, H, W, nullptr, bbBase,  0, BB_PB,  bbOff[s]);
    };

    // ---- backbone + heads (NHWC f16 throughout) ----
    run_conv(Xp, 0, 512, 512, A,  nullptr, 1, 0, 0);     // 3->16
    run_conv(A,  1, 512, 512, Bb, nullptr, 1, 0, 0);     // 16->16
    run_pool(Bb, A, 16, 512, 512);                       // -> 16@256
    run_conv(A,  2, 256, 256, Bb, nullptr, 1, 0, 0);     // 16->32
    run_conv(Bb, 3, 256, 256, A,  nullptr, 1, 0, 0);     // 32->32
    run_pool(A, Bb, 32, 256, 256);                       // -> 32@128
    run_conv(Bb, 4, 128, 128, A,  nullptr, 1, 0, 0);     // 32->64
    run_conv(A,  5, 128, 128, Bb, nullptr, 1, 0, 0);     // 64->64
    run_pool(Bb, A, 64, 128, 128);                       // -> 64@64   (feature 0)
    run_preds(A, 0, 64, 64);

    run_conv(A,  6, 64, 64, Bb, nullptr, 1, 0, 0);       // 64->128
    run_conv(Bb, 7, 64, 64, A,  nullptr, 1, 0, 0);       // 128->128
    run_pool(A, Bb, 128, 64, 64);                        // -> 128@32  (feature 1)
    run_preds(Bb, 1, 32, 32);

    run_conv(Bb, 8, 32, 32, A,  nullptr, 1, 0, 0);
    run_conv(A,  9, 32, 32, Bb, nullptr, 1, 0, 0);
    run_pool(Bb, A, 128, 32, 32);                        // -> 128@16  (feature 2)
    run_preds(A, 2, 16, 16);

    run_conv(A, 10, 16, 16, Bb, nullptr, 1, 0, 0);
    run_conv(Bb, 11, 16, 16, A, nullptr, 1, 0, 0);
    run_pool(A, Bb, 128, 16, 16);                        // -> 128@8   (feature 3)
    run_preds(Bb, 3, 8, 8);

    gmaxpool<<<dim3((NB * 128 + 255) / 256), dim3(256), 0, stream>>>(Bb, A, 128, 64, NB); // feature 4
    run_preds(A, 4, 1, 1);

    // ---- anchors ----
    static const int   ah[5]   = {64, 32, 16, 8, 1};
    static const int   arow[5] = {0, 16384, 20480, 21504, 21760};
    static const float as0[5]  = {0.2f, 0.37f, 0.54f, 0.71f, 0.88f};
    static const float as1[5]  = {0.272f, 0.447f, 0.619f, 0.79f, 0.961f};
    for (int s = 0; s < 5; ++s) {
        const int n = ah[s] * ah[s] * 4;
        anchors_k<<<dim3((n + 255) / 256), dim3(256), 0, stream>>>(
            outF + (size_t)arow[s] * 4, ah[s], ah[s], as0[s], as1[s]);
    }
}
